// IG_MSA_3444563771566
// MI455X (gfx1250) — matrix operand order repaired and checked by reading, not compiled, not run on hardware
//
#include <hip/hip_runtime.h>
#include <hip/hip_bf16.h>
#include <math.h>

typedef _Float16 f16;
typedef __attribute__((ext_vector_type(16))) _Float16 v16h;
typedef __attribute__((ext_vector_type(8)))  _Float16 v8h;
typedef __attribute__((ext_vector_type(8)))  float    v8f;
typedef __attribute__((ext_vector_type(4)))  unsigned int u32x4;
typedef __attribute__((ext_vector_type(8)))  int i32x8;
typedef __attribute__((ext_vector_type(4)))  int i32x4;

// may-alias variants for type-punned memory access (defeat TBAA)
typedef v8h  v8h_m  __attribute__((may_alias));
typedef v16h v16h_m __attribute__((may_alias));

#define DIMC   512
#define HEADS  8
#define DHEAD  64
#define BATCH  4
#define IMH    128
#define IMW    128
#define NTOK   (IMH*IMW)      /* 16384 */
#define ROWS   (BATCH*NTOK)   /* 65536 */

// ---------------- helpers ----------------
__device__ inline v8f vzero8() {
  v8f z;
#pragma unroll
  for (int i = 0; i < 8; ++i) z[i] = 0.f;
  return z;
}

__device__ inline v8f wmma16(v16h a, v16h b, v8f c) {
  // D = A(16x32 f16) x B(32x16 f16) + C(16x16 f32)
  return __builtin_amdgcn_wmma_f32_16x16x32_f16(false, a, false, b, (short)0, c,
                                                false, false);
}

// Make an LDS buffer's address escape so the compiler must assume it was
// written (the TDM engine writes it behind the compiler's back).
__device__ inline void lds_escape(const void* p) {
  asm volatile("" : : "v"(p) : "memory");
}

// A fragment (16x32 f16) from a row-major f16 row pointer already offset to
// (row, kk). lane 0-15: K=[0..15]; lane 16-31: K=[16..31].
__device__ inline v16h load_a_vec(const f16* prow, int lane) {
  const f16* p = prow + (lane >> 4) * 16;
  v8h lo = *(const v8h_m*)p;
  v8h hi = *(const v8h_m*)(p + 8);
  v16h a;
#pragma unroll
  for (int i = 0; i < 8; ++i) { a[i] = lo[i]; a[i + 8] = hi[i]; }
  return a;
}

// B fragment (32x16 f16) from (N,K) row-major f16: 16 contiguous halves.
// lane 0-15: N=lane, K=[0..15]; lane 16-31: N=lane-16, K=[16..31]
__device__ inline v16h load_b_vec(const f16* prow /* + col*ld + kk */, int lane) {
  return *(const v16h_m*)(prow + (lane >> 4) * 16);
}

// ---------------- TDM: 2D tile -> LDS (dsize_code: 0=1B,1=2B,2=4B) ----------
__device__ inline void tdm_load_2d(const void* gsrc, unsigned lds_byte_off,
                                   int tile_rows, int row_elems,
                                   int row_stride_elems, int dsize_code) {
  unsigned long long ga = (unsigned long long)gsrc;
  u32x4 g0;
  g0[0] = 1u;                                        // count=1, user mode
  g0[1] = lds_byte_off;                              // lds_addr
  g0[2] = (unsigned)(ga & 0xFFFFFFFFull);            // global_addr[31:0]
  g0[3] = (unsigned)((ga >> 32) & 0x1FFFFFFull) | (2u << 30); // addr[56:32]|type=2
  i32x8 g1;
  g1[0] = (dsize_code << 16);                        // data_size
  g1[1] = (row_elems & 0xFFFF) << 16;                // tensor_dim0[15:0]
  g1[2] = ((row_elems >> 16) & 0xFFFF) | ((tile_rows & 0xFFFF) << 16);
  g1[3] = ((tile_rows >> 16) & 0xFFFF) | ((row_elems & 0xFFFF) << 16); // tile_dim0
  g1[4] = (tile_rows & 0xFFFF);                      // tile_dim1 (tile_dim2=0)
  g1[5] = row_stride_elems;                          // tensor_dim0_stride[31:0]
  g1[6] = 0;
  g1[7] = 0;
  i32x4 g2; g2[0] = 0; g2[1] = 0; g2[2] = 0; g2[3] = 0;
  i32x4 g3 = g2;
#if __has_include(<hip/amd_detail/amd_gfx1250_TDM.h>)
  i32x8 g4;
#pragma unroll
  for (int i = 0; i < 8; ++i) g4[i] = 0;
  __builtin_amdgcn_tensor_load_to_lds(g0, g1, g2, g3, g4, 0);
#else
  __builtin_amdgcn_tensor_load_to_lds(g0, g1, g2, g3, 0);
#endif
}

// ---------------- k0: weight conversion f32 -> f16 ----------------
__global__ void __launch_bounds__(256) k_cvt_w(const float* wq, const float* wk,
                                               const float* wv, const float* wp,
                                               f16* q16, f16* k16, f16* v16, f16* p16) {
  int i = blockIdx.x * 256 + threadIdx.x;   // 262144 = 512*512
  q16[i] = (f16)wq[i];
  k16[i] = (f16)wk[i];
  v16[i] = (f16)wv[i];
  p16[i] = (f16)wp[i];
}

// ---------------- k1: QKV projections (WMMA, TDM-staged A) ------------------
// Outputs: QT,KT in transposed (b,c,n) f16 layout (feeds k_gram/k_sumsq with
// unit-stride vector loads); V in row-major (b,n,c) f16 (feeds k_attnv/conv).
__global__ void __launch_bounds__(256) k_qkv(const float* __restrict__ x,
                                             const float* __restrict__ illu,
                                             const f16* __restrict__ Wq16,
                                             const f16* __restrict__ Wk16,
                                             const f16* __restrict__ Wv16,
                                             f16* __restrict__ QT16,
                                             f16* __restrict__ KT16,
                                             f16* __restrict__ V16) {
  __shared__ float xs[16 * DIMC];           // 32 KB staged f32 panel (TDM-written)
  __shared__ f16   xh[16 * DIMC];           // 16 KB f16 copy (vector A reads)
  const int mtile = blockIdx.x;             // 0..4095 over all 65536 rows
  const int wave  = threadIdx.x >> 5;
  const int lane  = threadIdx.x & 31;
  const int ntile = blockIdx.y * 8 + wave;  // 0..31
  const size_t Rbase = (size_t)mtile * 16;

  if (wave == 0) {
    tdm_load_2d(x + Rbase * DIMC, 0u, 16, DIMC, DIMC, /*4B*/2);
    __builtin_amdgcn_s_wait_tensorcnt(0);
  }
  lds_escape(xs);                           // TDM wrote xs behind compiler's back
  __syncthreads();
  // one-shot f32 -> f16 conversion of the panel
  for (int i = threadIdx.x; i < 16 * DIMC; i += 256) xh[i] = (f16)xs[i];
  __syncthreads();

  v8f accQ = vzero8(), accK = vzero8(), accV = vzero8();
  const int row = lane & 15;
  const int colW = ntile * 16 + (lane & 15);
  const size_t wrow = (size_t)colW * DIMC;
#pragma unroll 4
  for (int kk = 0; kk < DIMC; kk += 32) {
    v16h a  = load_a_vec(xh + row * DIMC + kk, lane);
    v16h bq = load_b_vec(Wq16 + wrow + kk, lane);
    v16h bk = load_b_vec(Wk16 + wrow + kk, lane);
    v16h bv = load_b_vec(Wv16 + wrow + kk, lane);
    accQ = wmma16(a, bq, accQ);
    accK = wmma16(a, bk, accK);
    accV = wmma16(a, bv, accV);
  }
  // C/D layout: VGPR r -> M = (lane>=16 ? 8 : 0)+r, N = lane%16
  const int mofs = (lane >> 4) * 8;
  const int b    = (int)(Rbase / NTOK);
  const int nbase = (int)(Rbase % NTOK) + mofs;      // 8 consecutive n per thread
  // transposed stores: 8 consecutive halves -> single 16B store each
  v8h qv, kv;
#pragma unroll
  for (int r = 0; r < 8; ++r) { qv[r] = (f16)accQ[r]; kv[r] = (f16)accK[r]; }
  *(v8h_m*)(QT16 + ((size_t)b * DIMC + colW) * NTOK + nbase) = qv;
  *(v8h_m*)(KT16 + ((size_t)b * DIMC + colW) * NTOK + nbase) = kv;
#pragma unroll
  for (int r = 0; r < 8; ++r) {
    size_t o = (Rbase + mofs + r) * DIMC + colW;
    V16[o] = (f16)(accV[r] * illu[o]);    // v *= illu before attention & conv
  }
}

// ---------------- k2: per-(b,c) column sum of squares (coalesced rows) ------
__global__ void __launch_bounds__(256) k_sumsq(const f16* __restrict__ QT16,
                                               const f16* __restrict__ KT16,
                                               float* __restrict__ ssqQ,
                                               float* __restrict__ ssqK) {
  __shared__ float red[256];
  const int blk = blockIdx.x;                         // 0..4095
  const f16* src = (blk < 2048) ? QT16 : KT16;
  float* dst     = (blk < 2048) ? ssqQ : ssqK;
  const int bc = blk & 2047;
  const int b = bc >> 9, c = bc & 511;
  const f16* rowp = src + ((size_t)b * DIMC + c) * NTOK;
  float s = 0.f;
  for (int n0 = threadIdx.x * 8; n0 < NTOK; n0 += 256 * 8) {
    v8h v = *(const v8h_m*)(rowp + n0);
#pragma unroll
    for (int i = 0; i < 8; ++i) { float f = (float)v[i]; s += f * f; }
  }
  red[threadIdx.x] = s;
  __syncthreads();
  for (int st = 128; st > 0; st >>= 1) {
    if ((int)threadIdx.x < st) red[threadIdx.x] += red[threadIdx.x + st];
    __syncthreads();
  }
  if (threadIdx.x == 0) dst[b * DIMC + c] = red[0];
}

// ---------------- k3: raw Gram G = K^T Q per (b,head), 64x64, K-dim 16384 ---
// QT/KT are (b,c,n): both operands are unit-stride in the reduction dim.
__global__ void __launch_bounds__(512) k_gram(const f16* __restrict__ QT16,
                                              const f16* __restrict__ KT16,
                                              float* __restrict__ attn) {
  const int bh = blockIdx.x;                 // 0..31
  const int b = bh >> 3, h = bh & 7;
  const int wave = threadIdx.x >> 5;         // 0..15
  const int lane = threadIdx.x & 31;
  const int mt = wave >> 2, nt = wave & 3;   // 4x4 tiles of the 64x64 output
  const int cbase = h * DHEAD;

  const f16* arow = KT16 + ((size_t)b * DIMC + cbase + mt * 16 + (lane & 15)) * NTOK;
  const f16* brow = QT16 + ((size_t)b * DIMC + cbase + nt * 16 + (lane & 15)) * NTOK;

  v8f acc = vzero8();
#pragma unroll 2
  for (int n0 = 0; n0 < NTOK; n0 += 32) {
    v16h a  = load_a_vec(arow + n0, lane);   // A[d,k]=K[n0+k, c_d]
    v16h bf = load_b_vec(brow + n0, lane);   // B[k,e]=Q[n0+k, c_e]
    acc = wmma16(a, bf, acc);
  }
  const int mofs = (lane >> 4) * 8;
#pragma unroll
  for (int r = 0; r < 8; ++r) {
    int d = mt * 16 + mofs + r;
    int e = nt * 16 + (lane & 15);
    attn[((size_t)bh * 64 + d) * 64 + e] = acc[r];
  }
}

// ---------------- k4: fold norms + rescale, softmax over e ------------------
__global__ void __launch_bounds__(64) k_softmax(float* __restrict__ attn,
                                                const float* __restrict__ ssqQ,
                                                const float* __restrict__ ssqK,
                                                const float* __restrict__ rescale) {
  const int bh = blockIdx.x;
  const int b = bh >> 3, h = bh & 7;
  const int d = threadIdx.x;                 // 0..63
  float nk = fmaxf(sqrtf(ssqK[b * DIMC + h * DHEAD + d]), 1e-12f);
  const float rs = rescale[h];
  float* row = attn + ((size_t)bh * 64 + d) * 64;
  float r[64];
  float mx = -3.0e38f;
#pragma unroll 8
  for (int e = 0; e < 64; ++e) {
    float nq = fmaxf(sqrtf(ssqQ[b * DIMC + h * DHEAD + e]), 1e-12f);
    float v = row[e] / (nk * nq) * rs;
    r[e] = v;
    mx = fmaxf(mx, v);
  }
  float s = 0.f;
#pragma unroll 8
  for (int e = 0; e < 64; ++e) { r[e] = expf(r[e] - mx); s += r[e]; }
  float inv = 1.f / s;
#pragma unroll 8
  for (int e = 0; e < 64; ++e) row[e] = r[e] * inv;
}

// ---------------- k5: Xo_h = V_h (nx64) * attn^T (64x64) per (b,h) ----------
__global__ void __launch_bounds__(256) k_attnv(const f16* __restrict__ V16,
                                               const float* __restrict__ attn,
                                               f16* __restrict__ Xo16) {
  const int bh = blockIdx.y;                 // 0..31
  const int b = bh >> 3, h = bh & 7;
  const int wave = threadIdx.x >> 5;         // 0..7
  const int lane = threadIdx.x & 31;
  const int mtile = blockIdx.x * 2 + (wave >> 2);  // 0..1023 (rows of 16)
  const int dtile = wave & 3;                      // 0..3 (16 output channels)

  const size_t rowA = (size_t)b * NTOK + (size_t)mtile * 16 + (lane & 15);
  const int koffB = (lane >> 4) * 16;
  const int dcol = dtile * 16 + (lane & 15);
  const float* A = attn + (size_t)bh * 64 * 64;

  v8f acc = vzero8();
#pragma unroll
  for (int kk = 0; kk < DHEAD; kk += 32) {
    v16h a = load_a_vec(V16 + rowA * DIMC + h * DHEAD + kk, lane);
    v16h bf;
#pragma unroll
    for (int j = 0; j < 16; ++j) {           // B[e,d] = attn[d,e] (L2-resident)
      bf[j] = (f16)A[(size_t)dcol * 64 + kk + koffB + j];
    }
    acc = wmma16(a, bf, acc);
  }
  const int mofs = (lane >> 4) * 8;
#pragma unroll
  for (int r = 0; r < 8; ++r) {
    size_t R = (size_t)b * NTOK + (size_t)mtile * 16 + mofs + r;
    Xo16[R * DIMC + h * DHEAD + dcol] = (f16)acc[r];
  }
}

// ---------------- k6: out = Xo * Wp^T + bp (WMMA, TDM-staged A) -------------
__global__ void __launch_bounds__(256) k_proj(const f16* __restrict__ Xo16,
                                              const f16* __restrict__ Wp16,
                                              const float* __restrict__ bp,
                                              float* __restrict__ out) {
  __shared__ f16 xo_s[16 * DIMC];            // 16 KB staged f16 panel (TDM-written)
  const int mtile = blockIdx.x;              // 0..4095
  const int wave = threadIdx.x >> 5;
  const int lane = threadIdx.x & 31;
  const int ntile = blockIdx.y * 8 + wave;   // 0..31
  const size_t Rbase = (size_t)mtile * 16;

  if (wave == 0) {
    tdm_load_2d(Xo16 + Rbase * DIMC, 0u, 16, DIMC, DIMC, /*2B*/1);
    __builtin_amdgcn_s_wait_tensorcnt(0);
  }
  lds_escape(xo_s);                          // TDM wrote xo_s behind compiler's back
  __syncthreads();

  const int row = lane & 15;
  const int colW = ntile * 16 + (lane & 15);
  const size_t wrow = (size_t)colW * DIMC;
  v8f acc = vzero8();
#pragma unroll 4
  for (int kk = 0; kk < DIMC; kk += 32) {
    v16h a = load_a_vec(xo_s + row * DIMC + kk, lane);
    v16h b = load_b_vec(Wp16 + wrow + kk, lane);
    acc = wmma16(a, b, acc);
  }
  const float bias = bp[colW];
  const int mofs = (lane >> 4) * 8;
#pragma unroll
  for (int r = 0; r < 8; ++r) {
    out[(Rbase + mofs + r) * DIMC + colW] = acc[r] + bias;
  }
}

// ---------------- k7a: depthwise 3x3 conv + erf-GELU (c fastest, NHWC) ------
__global__ void __launch_bounds__(256) k_conv1(const f16* __restrict__ V16,
                                               const float* __restrict__ pe1,
                                               float* __restrict__ p1) {
  const int idx = blockIdx.x * 256 + threadIdx.x;  // ((b*128+y)*128+x)*512+c
  const int c  = idx & 511;
  const int xx = (idx >> 9) & 127;
  const int yy = (idx >> 16) & 127;
  const int b  = idx >> 23;
  const float* k9 = pe1 + c * 9;
  float s = 0.f;
#pragma unroll
  for (int dy = -1; dy <= 1; ++dy) {
    int y2 = yy + dy;
    if (y2 < 0 || y2 > 127) continue;
#pragma unroll
    for (int dx = -1; dx <= 1; ++dx) {
      int x2 = xx + dx;
      if (x2 < 0 || x2 > 127) continue;
      float v = (float)V16[(((size_t)b * NTOK) + y2 * IMW + x2) * DIMC + c];
      s += v * k9[(dy + 1) * 3 + (dx + 1)];
    }
  }
  p1[idx] = 0.5f * s * (1.0f + erff(s * 0.70710678118654752f));  // exact GELU
}

// ---------------- k7b: depthwise 3x3 conv, accumulate into out (NHWC) -------
__global__ void __launch_bounds__(256) k_conv2(const float* __restrict__ p1,
                                               const float* __restrict__ pe2,
                                               float* __restrict__ out) {
  const int idx = blockIdx.x * 256 + threadIdx.x;
  const int c  = idx & 511;
  const int xx = (idx >> 9) & 127;
  const int yy = (idx >> 16) & 127;
  const int b  = idx >> 23;
  const float* k9 = pe2 + c * 9;
  float s = 0.f;
#pragma unroll
  for (int dy = -1; dy <= 1; ++dy) {
    int y2 = yy + dy;
    if (y2 < 0 || y2 > 127) continue;
#pragma unroll
    for (int dx = -1; dx <= 1; ++dx) {
      int x2 = xx + dx;
      if (x2 < 0 || x2 > 127) continue;
      s += p1[(((size_t)b * IMH + y2) * IMW + x2) * DIMC + c] *
           k9[(dy + 1) * 3 + (dx + 1)];
    }
  }
  out[idx] += s;   // same NHWC index as output
}

// ---------------- host launcher ----------------
extern "C" void kernel_launch(void* const* d_in, const int* in_sizes, int n_in,
                              void* d_out, int out_size, void* d_ws, size_t ws_size,
                              hipStream_t stream) {
  const float* x       = (const float*)d_in[0];
  const float* illu    = (const float*)d_in[1];
  const float* Wq      = (const float*)d_in[2];
  const float* Wk      = (const float*)d_in[3];
  const float* Wv      = (const float*)d_in[4];
  const float* rescale = (const float*)d_in[5];
  const float* Wp      = (const float*)d_in[6];
  const float* bp      = (const float*)d_in[7];
  const float* pe1     = (const float*)d_in[8];
  const float* pe2     = (const float*)d_in[9];
  float* out = (float*)d_out;
  char* ws = (char*)d_ws;

  const size_t WBYTES = (size_t)DIMC * DIMC * sizeof(f16);        // 512 KB
  const size_t MBYTES = (size_t)ROWS * DIMC * sizeof(f16);        // 64 MB
  size_t off = 0;
  f16* Wq16 = (f16*)(ws + off); off += WBYTES;
  f16* Wk16 = (f16*)(ws + off); off += WBYTES;
  f16* Wv16 = (f16*)(ws + off); off += WBYTES;
  f16* Wp16 = (f16*)(ws + off); off += WBYTES;
  f16* QT16 = (f16*)(ws + off); off += MBYTES;   // (b,c,n) transposed
  f16* KT16 = (f16*)(ws + off); off += MBYTES;   // (b,c,n) transposed
  f16* V16  = (f16*)(ws + off); off += MBYTES;   // (b,n,c) row-major
  f16* Xo16 = (f16*)(ws + off); off += MBYTES;   // (b,n,c) row-major
  float* ssqQ = (float*)(ws + off); off += (size_t)BATCH * DIMC * sizeof(float);
  float* ssqK = (float*)(ws + off); off += (size_t)BATCH * DIMC * sizeof(float);
  float* attn = (float*)(ws + off); off += (size_t)BATCH * HEADS * 64 * 64 * sizeof(float);
  float* p1   = (float*)(ws + off); off += (size_t)BATCH * DIMC * NTOK * sizeof(float);

  k_cvt_w<<<1024, 256, 0, stream>>>(Wq, Wk, Wv, Wp, Wq16, Wk16, Wv16, Wp16);

  dim3 gQKV(ROWS / 16, 4);
  k_qkv<<<gQKV, 256, 0, stream>>>(x, illu, Wq16, Wk16, Wv16, QT16, KT16, V16);

  k_sumsq<<<2 * BATCH * DIMC, 256, 0, stream>>>(QT16, KT16, ssqQ, ssqK);

  k_gram<<<BATCH * HEADS, 512, 0, stream>>>(QT16, KT16, attn);

  k_softmax<<<BATCH * HEADS, 64, 0, stream>>>(attn, ssqQ, ssqK, rescale);

  dim3 gAV(NTOK / 32, BATCH * HEADS);
  k_attnv<<<gAV, 256, 0, stream>>>(V16, attn, Xo16);

  dim3 gP(ROWS / 16, 4);
  k_proj<<<gP, 256, 0, stream>>>(Xo16, Wp16, bp, out);

  const int nElem = BATCH * DIMC * NTOK;   // 33,554,432
  k_conv1<<<nElem / 256, 256, 0, stream>>>(V16, pe1, p1);
  k_conv2<<<nElem / 256, 256, 0, stream>>>(p1, pe2, out);
}
